// ContrastiveLoss_V1_841813590238
// MI455X (gfx1250) — matrix operand order repaired and checked by reading, not compiled, not run on hardware
//
#include <hip/hip_runtime.h>
#include <hip/hip_bf16.h>

typedef __attribute__((ext_vector_type(16))) _Float16 v16h;
typedef __attribute__((ext_vector_type(8)))  _Float16 v8h;
typedef __attribute__((ext_vector_type(8)))  float    v8f;

#define N_ROWS 4096
#define DIM    1024
#define EPS_PD 1e-6f
#define EPS_NORM 1e-12f

// ---------------------------------------------------------------------------
// Kernel 1: per-row L2 normalize; emit f16 hi/lo split + r_i = ||e||^2, s_i = sum(e)
// One 256-thread block per row; each thread handles 4 consecutive floats.
// ---------------------------------------------------------------------------
__global__ void cl_normalize_split(const float* __restrict__ X,
                                   _Float16* __restrict__ Ehi,
                                   _Float16* __restrict__ Elo,
                                   float* __restrict__ r,
                                   float* __restrict__ s) {
    const int row = blockIdx.x;
    const int tid = threadIdx.x;              // 0..255
    const float4 v = ((const float4*)(X + (size_t)row * DIM))[tid];

    float sq = v.x*v.x + v.y*v.y + v.z*v.z + v.w*v.w;
    float sm = v.x + v.y + v.z + v.w;

    __shared__ float sSq[256];
    __shared__ float sSm[256];
    sSq[tid] = sq; sSm[tid] = sm;
    __syncthreads();
    #pragma unroll
    for (int off = 128; off > 0; off >>= 1) {
        if (tid < off) { sSq[tid] += sSq[tid + off]; sSm[tid] += sSm[tid + off]; }
        __syncthreads();
    }
    const float sumsq = sSq[0];
    const float sumx  = sSm[0];

    const float inv = 1.0f / fmaxf(sqrtf(sumsq), EPS_NORM);

    float e[4] = { v.x * inv, v.y * inv, v.z * inv, v.w * inv };
    _Float16* ph = Ehi + (size_t)row * DIM + tid * 4;
    _Float16* pl = Elo + (size_t)row * DIM + tid * 4;
    #pragma unroll
    for (int j = 0; j < 4; ++j) {
        _Float16 hi = (_Float16)e[j];
        _Float16 lo = (_Float16)(e[j] - (float)hi);
        ph[j] = hi;
        pl[j] = lo;
    }
    if (tid == 0) {
        r[row] = sumsq * inv * inv;   // == ||e||^2
        s[row] = sumx * inv;          // == sum(e)
    }
}

// A fragment (16x32 f16, MxK): lane holds row m0+(lane&15);
// halves [k + ah .. +8) in VGPR0-3, [k + ah + 8 .. +8) in VGPR4-7, ah = lane<16?0:16.
static __device__ __forceinline__ v16h load_a_frag(const _Float16* p) {
    v8h x0 = *(const v8h*)(p);
    v8h x1 = *(const v8h*)(p + 8);
    return __builtin_shufflevector(x0, x1,
        0,1,2,3,4,5,6,7,8,9,10,11,12,13,14,15);
}
// B fragment (32x16 f16, KxN): lane holds column n0+(lane&15);
// 16 contiguous halves at k + (lane<16?0:16).
static __device__ __forceinline__ v16h load_b_frag(const _Float16* p) {
    return *(const v16h*)p;
}

static __device__ __forceinline__ v8f wmma3(v16h ah, v16h al, v16h bh, v16h bl, v8f c) {
    // g ~= hi*hi + hi*lo + lo*hi  (lo*lo term negligible: ~2^-22 relative)
    c = __builtin_amdgcn_wmma_f32_16x16x32_f16(false, ah, false, bh, (short)0, c, false, false);
    c = __builtin_amdgcn_wmma_f32_16x16x32_f16(false, ah, false, bl, (short)0, c, false, false);
    c = __builtin_amdgcn_wmma_f32_16x16x32_f16(false, al, false, bh, (short)0, c, false, false);
    return c;
}

// ---------------------------------------------------------------------------
// Kernel 2: tiled Gram + fused contrastive epilogue.
// Block = 128 threads = 4 waves, arranged 2x2; each wave owns a 32x32 macro
// tile (2x2 WMMA accumulators, A-fragments reused across the N-subtiles).
// 12 v_wmma per 32-wide K step per wave. One float partial per block.
// ---------------------------------------------------------------------------
__global__ void cl_gram_loss(const _Float16* __restrict__ Ehi,
                             const _Float16* __restrict__ Elo,
                             const float* __restrict__ r,
                             const float* __restrict__ s,
                             float* __restrict__ partial) {
    const int lane = threadIdx.x & 31;
    const int wave = threadIdx.x >> 5;        // 0..3
    const int lm   = lane & 15;
    const int ah   = (lane < 16) ? 0 : 16;    // A fragment K sub-offset
    const int bh   = (lane < 16) ? 0 : 16;    // B fragment K sub-offset

    const int wm0 = blockIdx.y * 64 + (wave >> 1) * 32;   // rows of tile
    const int wn0 = blockIdx.x * 64 + (wave & 1) * 32;    // cols of tile

    // Per-lane base pointers (rows of E for both A and B fragments).
    const size_t rA0 = (size_t)(wm0 + lm) * DIM + ah;
    const size_t rA1 = rA0 + (size_t)16 * DIM;
    const size_t rB0 = (size_t)(wn0 + lm) * DIM + bh;
    const size_t rB1 = rB0 + (size_t)16 * DIM;

    v8f c00 = {}; v8f c01 = {}; v8f c10 = {}; v8f c11 = {};

    for (int k = 0; k < DIM; k += 32) {
        v16h a0h = load_a_frag(Ehi + rA0 + k);
        v16h a0l = load_a_frag(Elo + rA0 + k);
        v16h a1h = load_a_frag(Ehi + rA1 + k);
        v16h a1l = load_a_frag(Elo + rA1 + k);
        v16h b0h = load_b_frag(Ehi + rB0 + k);
        v16h b0l = load_b_frag(Elo + rB0 + k);
        v16h b1h = load_b_frag(Ehi + rB1 + k);
        v16h b1l = load_b_frag(Elo + rB1 + k);

        c00 = wmma3(a0h, a0l, b0h, b0l, c00);
        c01 = wmma3(a0h, a0l, b1h, b1l, c01);
        c10 = wmma3(a1h, a1l, b0h, b0l, c10);
        c11 = wmma3(a1h, a1l, b1h, b1l, c11);
    }

    // Epilogue: d2 = r_m + r_n - 2 g + 2 eps (s_m - s_n) + D eps^2 ; clamp; mask diag.
    const float deps2 = (float)DIM * EPS_PD * EPS_PD;
    float acc = 0.0f;
    const v8f* cmat[2][2] = { { &c00, &c01 }, { &c10, &c11 } };

    #pragma unroll
    for (int mi = 0; mi < 2; ++mi) {
        const int mbase = wm0 + mi * 16 + ((lane < 16) ? 0 : 8);
        float rm[8], sm[8];
        #pragma unroll
        for (int v = 0; v < 8; ++v) { rm[v] = r[mbase + v]; sm[v] = s[mbase + v]; }
        #pragma unroll
        for (int ni = 0; ni < 2; ++ni) {
            const int col = wn0 + ni * 16 + lm;
            const float rc = r[col];
            const float sc = s[col];
            const v8f g = *cmat[mi][ni];
            #pragma unroll
            for (int v = 0; v < 8; ++v) {
                const int row = mbase + v;
                float d2 = rm[v] + rc - 2.0f * g[v]
                         + 2.0f * EPS_PD * (sm[v] - sc) + deps2;
                d2 = fmaxf(d2, 0.0f);
                acc += (row == col) ? 0.0f : d2;
            }
        }
    }

    // Wave reduction (wave32), then block reduction via LDS.
    #pragma unroll
    for (int off = 16; off > 0; off >>= 1)
        acc += __shfl_xor(acc, off, 32);

    __shared__ float sWave[4];
    if (lane == 0) sWave[wave] = acc;
    __syncthreads();
    if (threadIdx.x == 0) {
        float t = sWave[0] + sWave[1] + sWave[2] + sWave[3];
        partial[blockIdx.y * gridDim.x + blockIdx.x] = t;
    }
}

// ---------------------------------------------------------------------------
// Kernel 3: deterministic fixed-order reduction of block partials -> loss.
// ---------------------------------------------------------------------------
__global__ void cl_finalize(const float* __restrict__ partial, int nPart,
                            float* __restrict__ out) {
    __shared__ float sh[256];
    float a = 0.0f;
    for (int i = threadIdx.x; i < nPart; i += 256) a += partial[i];
    sh[threadIdx.x] = a;
    __syncthreads();
    #pragma unroll
    for (int off = 128; off > 0; off >>= 1) {
        if (threadIdx.x < off) sh[threadIdx.x] += sh[threadIdx.x + off];
        __syncthreads();
    }
    if (threadIdx.x == 0) {
        const float inv_ncomp = 1.0f / ((float)N_ROWS * (float)(N_ROWS - 1));
        out[0] = sh[0] * inv_ncomp;
    }
}

extern "C" void kernel_launch(void* const* d_in, const int* in_sizes, int n_in,
                              void* d_out, int out_size, void* d_ws, size_t ws_size,
                              hipStream_t stream) {
    (void)in_sizes; (void)n_in; (void)out_size; (void)ws_size;
    const float* X = (const float*)d_in[0];   // embeddings [4096,1024] f32
    // d_in[1] = labels (unused by the reference math)

    char* ws = (char*)d_ws;
    const size_t szE   = (size_t)N_ROWS * DIM * sizeof(_Float16); // 8 MiB each
    _Float16* Ehi     = (_Float16*)(ws);
    _Float16* Elo     = (_Float16*)(ws + szE);
    float*    r       = (float*)(ws + 2 * szE);
    float*    s       = (float*)(ws + 2 * szE + N_ROWS * sizeof(float));
    float*    partial = (float*)(ws + 2 * szE + 2 * N_ROWS * sizeof(float));

    cl_normalize_split<<<N_ROWS, 256, 0, stream>>>(X, Ehi, Elo, r, s);

    dim3 grid(N_ROWS / 64, N_ROWS / 64);   // 64 x 64 blocks, 64x64 tile each
    cl_gram_loss<<<grid, 128, 0, stream>>>(Ehi, Elo, r, s, partial);

    const int nPart = (N_ROWS / 64) * (N_ROWS / 64);  // 4096
    cl_finalize<<<1, 256, 0, stream>>>(partial, nPart, (float*)d_out);
}
